// MultiHeadedDotProductSelfAttention_30846455120624
// MI455X (gfx1250) — matrix operand order repaired and checked by reading, not compiled, not run on hardware
//
#include <hip/hip_runtime.h>

// ---------------------------------------------------------------------------
// Types for gfx1250 WMMA (wave32)
// ---------------------------------------------------------------------------
typedef __bf16 v16bf __attribute__((ext_vector_type(16)));
typedef float  v8f   __attribute__((ext_vector_type(8)));
typedef unsigned int u32x4 __attribute__((ext_vector_type(4)));
typedef unsigned int u32x8 __attribute__((ext_vector_type(8)));

union Frag16 {
    v16bf v;
    u32x4 q[2];
    unsigned short s[16];
};

__device__ __forceinline__ unsigned short f32_to_bf16(float f) {
    union { float f; unsigned u; } cv;
    cv.f = f;
    unsigned u = cv.u;
    unsigned r = u + 0x7FFFu + ((u >> 16) & 1u);  // round-to-nearest-even
    return (unsigned short)(r >> 16);
}

// lane-XOR swap via DS swizzle (immediate pattern, no index VGPR needed)
template <int MASK>
__device__ __forceinline__ float swz_xor_f(float x) {
    int v = __float_as_int(x);
    v = __builtin_amdgcn_ds_swizzle(v, (MASK << 10) | 0x1F);
    return __int_as_float(v);
}

static constexpr int BATCH  = 2;
static constexpr int SEQ    = 2048;
static constexpr int DMODEL = 2048;
static constexpr int HEADS  = 16;
static constexpr int DHEAD  = 128;
static constexpr int MROWS  = BATCH * SEQ;                 // 4096
static constexpr float QK_SCALE = 0.29730177875068026f;    // 128^(-1/4)

// ---------------------------------------------------------------------------
// fp32 -> bf16 bit conversion, 8 elements / thread, 16B stores
// ---------------------------------------------------------------------------
__global__ __launch_bounds__(256)
void f32_to_bf16_kernel(const float* __restrict__ src,
                        unsigned int* __restrict__ dst, int n8) {
    int i = blockIdx.x * 256 + threadIdx.x;
    if (i >= n8) return;
    const float* s = src + (size_t)i * 8;
    u32x4 out;
#pragma unroll
    for (int j = 0; j < 4; ++j) {
        unsigned lo16 = f32_to_bf16(s[2 * j]);
        unsigned hi16 = f32_to_bf16(s[2 * j + 1]);
        out[j] = lo16 | (hi16 << 16);
    }
    *(u32x4*)(dst + (size_t)i * 4) = out;
}

// ---------------------------------------------------------------------------
// Tensor Data Mover: DMA one 64(n) x 32(k) bf16 W tile into LDS.
// D# group0: count=1 | lds_addr | global_addr[56:0] | type=2.
// D# group1: data_size=4B, pad_enable (every 16 dwords insert 4 dwords ->
//            LDS row stride 80B, matching ushort[64][40]), tensor = 1024x2048
//            dwords, tile = 16x64 dwords, dim0 stride = 1024 dwords.
// Issued by one wave; completion via TENSORcnt.
// ---------------------------------------------------------------------------
__device__ __forceinline__ void tdm_load_w_tile(unsigned lds_addr,
                                                unsigned long long gaddr,
                                                u32x8 g1) {
    u32x4 g0;
    g0.x = 1u;                                   // count=1, user descriptor
    g0.y = lds_addr;                             // LDS byte address
    g0.z = (unsigned)gaddr;                      // global_addr[31:0]
    g0.w = (unsigned)(gaddr >> 32) | (2u << 30); // global_addr[56:32] | type=2
    asm volatile("tensor_load_to_lds %0, %1" :: "s"(g0), "s"(g1) : "memory");
}

// ---------------------------------------------------------------------------
// C = A(bf16, MxK) * W(bf16, NxK)^T + bias, optional scale.
// OMODE 0: bf16 row-major [M][N]        (Q, K with QK_SCALE)
// OMODE 1: bf16 transposed [b][h][d][s] (V)
// OMODE 2: f32 row-major  [M][N]        (final projection)
// Block: 256 threads = 8 waves; block tile 256(M) x 64(N); K step 32.
// Wave tile 32x64 = 8 WMMA per K-step; W tile double-buffered via TDM.
// ---------------------------------------------------------------------------
template <int OMODE>
__global__ __launch_bounds__(256)
void wmma_gemm_xwt(const unsigned short* __restrict__ Abf,
                   const unsigned short* __restrict__ Wbf,
                   const float* __restrict__ bias,
                   void* __restrict__ outp,
                   float scale) {
    constexpr int K = DMODEL;
    constexpr int N = DMODEL;
    __shared__ __attribute__((aligned(16))) unsigned short Wl[2][64][40];

    const int tid  = threadIdx.x;
    const int wv   = tid >> 5;
    const int lane = tid & 31;
    const int lo   = lane & 15;
    const int hi   = lane >> 4;
    const int m0   = blockIdx.x * 256 + wv * 32;
    const int n0   = blockIdx.y * 64;

    const v8f vz = {0.f, 0.f, 0.f, 0.f, 0.f, 0.f, 0.f, 0.f};
    v8f acc[2][4] = {{vz, vz, vz, vz}, {vz, vz, vz, vz}};

    // constant D# group 1 (see tdm_load_w_tile comment)
    const u32x8 g1 = {
        (2u << 16) | (1u << 20) | (3u << 22) | (3u << 25),  // dsz=4B, pad 16dw/+4dw
        (1024u << 16),   // tensor_dim0[15:0]=1024 dwords (row = 2048 bf16)
        (2048u << 16),   // tensor_dim0 hi=0 | tensor_dim1[15:0]=2048 rows
        (16u << 16),     // tensor_dim1 hi=0 | tile_dim0=16 dwords (32 bf16)
        64u,             // tile_dim1=64 rows, tile_dim2=0
        1024u,           // tensor_dim0_stride = 1024 dwords
        0u, 0u};
    const unsigned ldsb0 = (unsigned)(uintptr_t)&Wl[0][0][0];
    const unsigned ldsb1 = ldsb0 + 64u * 40u * 2u;
    const unsigned long long wbase =
        (unsigned long long)(uintptr_t)Wbf + (unsigned long long)n0 * K * 2u;

    if (wv == 0) {
        tdm_load_w_tile(ldsb0, wbase, g1);
        __builtin_amdgcn_s_wait_tensorcnt(0);
    }
    __syncthreads();

    for (int k0 = 0; k0 < K; k0 += 32) {
        const int  buf  = (k0 >> 5) & 1;
        const bool more = (k0 + 32) < K;
        if (wv == 0 && more)  // prefetch next W tile into the other buffer
            tdm_load_w_tile(buf ? ldsb0 : ldsb1,
                            wbase + (unsigned long long)(k0 + 32) * 2u, g1);

        // A fragments: lane lo = row m, interleaved k chunks (ISA A layout)
        Frag16 a0, a1;
        const unsigned short* ar0 = Abf + (size_t)(m0 + lo) * K + k0;
        const unsigned short* ar1 = ar0 + (size_t)16 * K;
        a0.q[0] = *(const u32x4*)(ar0 + 8 * hi);
        a0.q[1] = *(const u32x4*)(ar0 + 16 + 8 * hi);
        a1.q[0] = *(const u32x4*)(ar1 + 8 * hi);
        a1.q[1] = *(const u32x4*)(ar1 + 16 + 8 * hi);
        if (more) {  // stream A ahead of the consumer
            __builtin_prefetch(ar0 + 32, 0, 0);
            __builtin_prefetch(ar1 + 32, 0, 0);
        }

#pragma unroll
        for (int f = 0; f < 4; ++f) {
            // B fragment: lane lo = column n, interleaved k chunks (matches A)
            Frag16 b;
            const unsigned short* brow = &Wl[buf][f * 16 + lo][8 * hi];
            b.q[0] = *(const u32x4*)(brow);
            b.q[1] = *(const u32x4*)(brow + 16);
            acc[0][f] = __builtin_amdgcn_wmma_f32_16x16x32_bf16(
                false, a0.v, false, b.v, (short)0, acc[0][f], false, false);
            acc[1][f] = __builtin_amdgcn_wmma_f32_16x16x32_bf16(
                false, a1.v, false, b.v, (short)0, acc[1][f], false, false);
        }

        if (wv == 0 && more) __builtin_amdgcn_s_wait_tensorcnt(0);
        __syncthreads();
    }

#pragma unroll
    for (int g = 0; g < 2; ++g) {
        const int mg = m0 + g * 16;
#pragma unroll
        for (int f = 0; f < 4; ++f) {
            const int n = n0 + f * 16 + lo;
            const float bv = bias[n];
            if (OMODE == 1) {
                Frag16 pk;
#pragma unroll
                for (int r = 0; r < 8; ++r)
                    pk.s[r] = f32_to_bf16((acc[g][f][r] + bv) * scale);
                const int bidx = mg / SEQ;
                const int srow = (mg % SEQ) + 8 * hi;
                const int hh = n >> 7, dd = n & 127;
                unsigned short* dst = (unsigned short*)outp +
                    (size_t)((bidx * HEADS + hh) * DHEAD + dd) * SEQ + srow;
                *(u32x4*)dst = pk.q[0];
            } else {
#pragma unroll
                for (int r = 0; r < 8; ++r) {
                    const int m = mg + r + 8 * hi;
                    const float val = (acc[g][f][r] + bv) * scale;
                    if (OMODE == 0)
                        ((unsigned short*)outp)[(size_t)m * N + n] = f32_to_bf16(val);
                    else
                        ((float*)outp)[(size_t)m * N + n] = val;
                }
            }
        }
    }
}

// ---------------------------------------------------------------------------
// Flash attention: one wave per 16-query tile, online softmax, causal.
// Q,K: bf16 [b*S][D] (head-sliced); V: bf16 transposed [bh*128+d][S].
// Output: bf16 [b*S][D].
// ---------------------------------------------------------------------------
__global__ __launch_bounds__(128)
void flash_attn_kernel(const unsigned short* __restrict__ Qb,
                       const unsigned short* __restrict__ Kb,
                       const unsigned short* __restrict__ Vt,
                       unsigned short* __restrict__ A2) {
    __shared__ __attribute__((aligned(16))) unsigned short Pl[4][16][40];

    const int tid  = threadIdx.x;
    const int wv   = tid >> 5;
    const int lane = tid & 31;
    const int lo   = lane & 15;
    const int hi   = lane >> 4;
    const int bh   = blockIdx.y;          // 0..31
    const int b    = bh >> 4;
    const int hd0  = (bh & 15) * DHEAD;
    const int q0   = (blockIdx.x * 4 + wv) * 16;

    const float NEG_INF = -__builtin_inff();
    const v8f vz = {0.f, 0.f, 0.f, 0.f, 0.f, 0.f, 0.f, 0.f};

    // Q fragments (A layout): 16 queries x 128 dh
    Frag16 qf[4];
    {
        const unsigned short* qrow =
            Qb + (size_t)(b * SEQ + q0 + lo) * DMODEL + hd0;
#pragma unroll
        for (int i = 0; i < 4; ++i) {
            qf[i].q[0] = *(const u32x4*)(qrow + i * 32 + 8 * hi);
            qf[i].q[1] = *(const u32x4*)(qrow + i * 32 + 16 + 8 * hi);
        }
    }

    v8f o[8] = {vz, vz, vz, vz, vz, vz, vz, vz};  // 16 x 128 accumulator
    float mi[8], li[8];
#pragma unroll
    for (int r = 0; r < 8; ++r) { mi[r] = NEG_INF; li[r] = 0.f; }

    const int nt = ((q0 + 15) >> 5) + 1;  // causal: key tiles of 32
    for (int t = 0; t < nt; ++t) {
        const int kb = t * 32;

        // S = Q K^T : two 16x16 f32 tiles (keys kb..kb+15, kb+16..kb+31)
        v8f s0 = vz, s1 = vz;
        const unsigned short* kr0 =
            Kb + (size_t)(b * SEQ + kb + lo) * DMODEL + hd0;
        const unsigned short* kr1 = kr0 + (size_t)16 * DMODEL;
#pragma unroll
        for (int i = 0; i < 4; ++i) {
            Frag16 kf0, kf1;  // B layout: lane = key, interleaved dh k chunks (matches A)
            kf0.q[0] = *(const u32x4*)(kr0 + i * 32 + 8 * hi);
            kf0.q[1] = *(const u32x4*)(kr0 + i * 32 + 16 + 8 * hi);
            kf1.q[0] = *(const u32x4*)(kr1 + i * 32 + 8 * hi);
            kf1.q[1] = *(const u32x4*)(kr1 + i * 32 + 16 + 8 * hi);
            s0 = __builtin_amdgcn_wmma_f32_16x16x32_bf16(
                false, qf[i].v, false, kf0.v, (short)0, s0, false, false);
            s1 = __builtin_amdgcn_wmma_f32_16x16x32_bf16(
                false, qf[i].v, false, kf1.v, (short)0, s1, false, false);
        }

        // causal mask: only the diagonal (last) tile needs it
        if (t == nt - 1) {
#pragma unroll
            for (int r = 0; r < 8; ++r) {
                const int q = q0 + r + 8 * hi;
                if (kb + lo > q)      s0[r] = NEG_INF;
                if (kb + 16 + lo > q) s1[r] = NEG_INF;
            }
        }

        // online softmax (rows live across 16-lane halves; XOR swizzle reduce)
        float alpha[8];
#pragma unroll
        for (int r = 0; r < 8; ++r) {
            float mx = fmaxf(s0[r], s1[r]);
            mx = fmaxf(mx, swz_xor_f<1>(mx));
            mx = fmaxf(mx, swz_xor_f<2>(mx));
            mx = fmaxf(mx, swz_xor_f<4>(mx));
            mx = fmaxf(mx, swz_xor_f<8>(mx));
            const float mnew = fmaxf(mi[r], mx);
            const float p0 = __expf(s0[r] - mnew);
            const float p1 = __expf(s1[r] - mnew);
            float sum = p0 + p1;
            sum += swz_xor_f<1>(sum);
            sum += swz_xor_f<2>(sum);
            sum += swz_xor_f<4>(sum);
            sum += swz_xor_f<8>(sum);
            alpha[r] = __expf(mi[r] - mnew);
            li[r] = li[r] * alpha[r] + sum;
            mi[r] = mnew;
            // stage P (C layout) into LDS as [query][key] bf16
            Pl[wv][r + 8 * hi][lo]      = f32_to_bf16(p0);
            Pl[wv][r + 8 * hi][16 + lo] = f32_to_bf16(p1);
        }

        // rescale accumulator
#pragma unroll
        for (int j = 0; j < 8; ++j)
#pragma unroll
            for (int r = 0; r < 8; ++r)
                o[j][r] *= alpha[r];

        // cross-lane LDS round-trip: wait for ds stores (CDNA5 split counter)
        asm volatile("s_wait_dscnt 0" ::: "memory");

        // reload P as an A fragment (16 queries x 32 keys)
        Frag16 pf;
        const unsigned short* pr = &Pl[wv][lo][0];
        pf.q[0] = *(const u32x4*)(pr + 8 * hi);
        pf.q[1] = *(const u32x4*)(pr + 16 + 8 * hi);

        // O += P * V  (V^T rows contiguous over keys)
#pragma unroll
        for (int j = 0; j < 8; ++j) {
            Frag16 vf;
            const unsigned short* vrow =
                Vt + (size_t)(bh * DHEAD + j * 16 + lo) * SEQ + kb;
            vf.q[0] = *(const u32x4*)(vrow + 8 * hi);
            vf.q[1] = *(const u32x4*)(vrow + 16 + 8 * hi);
            o[j] = __builtin_amdgcn_wmma_f32_16x16x32_bf16(
                false, pf.v, false, vf.v, (short)0, o[j], false, false);
        }
    }

    // normalize and store as bf16 [b*S][D]
    float inv[8];
#pragma unroll
    for (int r = 0; r < 8; ++r) inv[r] = 1.f / li[r];
#pragma unroll
    for (int j = 0; j < 8; ++j)
#pragma unroll
        for (int r = 0; r < 8; ++r) {
            const int m = b * SEQ + q0 + r + 8 * hi;
            A2[(size_t)m * DMODEL + hd0 + j * 16 + lo] =
                f32_to_bf16(o[j][r] * inv[r]);
        }
}

// ---------------------------------------------------------------------------
// Host launch
// ---------------------------------------------------------------------------
extern "C" void kernel_launch(void* const* d_in, const int* in_sizes, int n_in,
                              void* d_out, int out_size, void* d_ws, size_t ws_size,
                              hipStream_t stream) {
    (void)in_sizes; (void)n_in; (void)out_size; (void)ws_size;

    const float* X  = (const float*)d_in[0];
    const float* Wq = (const float*)d_in[1];
    const float* bq = (const float*)d_in[2];
    const float* Wk = (const float*)d_in[3];
    const float* bk = (const float*)d_in[4];
    const float* Wv = (const float*)d_in[5];
    const float* bv = (const float*)d_in[6];
    const float* Wo = (const float*)d_in[7];
    const float* bo = (const float*)d_in[8];

    const size_t nX = (size_t)MROWS * DMODEL;   // 8,388,608
    const size_t nW = (size_t)DMODEL * DMODEL;  // 4,194,304

    unsigned short* Xb  = (unsigned short*)d_ws;
    unsigned short* Wqb = Xb  + nX;
    unsigned short* Wkb = Wqb + nW;
    unsigned short* Wvb = Wkb + nW;
    unsigned short* Wob = Wvb + nW;
    unsigned short* Qb  = Wob + nW;
    unsigned short* Kb  = Qb  + nX;
    unsigned short* Vtb = Kb  + nX;
    unsigned short* A2b = Vtb + nX;

    // 1) convert inputs to bf16 (8 elems / thread)
    f32_to_bf16_kernel<<<(int)(nX / 8 / 256), 256, 0, stream>>>(
        X, (unsigned int*)Xb, (int)(nX / 8));
    f32_to_bf16_kernel<<<(int)(nW / 8 / 256), 256, 0, stream>>>(
        Wq, (unsigned int*)Wqb, (int)(nW / 8));
    f32_to_bf16_kernel<<<(int)(nW / 8 / 256), 256, 0, stream>>>(
        Wk, (unsigned int*)Wkb, (int)(nW / 8));
    f32_to_bf16_kernel<<<(int)(nW / 8 / 256), 256, 0, stream>>>(
        Wv, (unsigned int*)Wvb, (int)(nW / 8));
    f32_to_bf16_kernel<<<(int)(nW / 8 / 256), 256, 0, stream>>>(
        Wo, (unsigned int*)Wob, (int)(nW / 8));

    // 2) Q/K/V projections (Q,K pre-scaled by 128^-0.25; V stored transposed)
    dim3 ggrid(MROWS / 256, DMODEL / 64);  // (16, 32)
    wmma_gemm_xwt<0><<<ggrid, 256, 0, stream>>>(Xb, Wqb, bq, Qb,  QK_SCALE);
    wmma_gemm_xwt<0><<<ggrid, 256, 0, stream>>>(Xb, Wkb, bk, Kb,  QK_SCALE);
    wmma_gemm_xwt<1><<<ggrid, 256, 0, stream>>>(Xb, Wvb, bv, Vtb, 1.0f);

    // 3) causal flash attention
    flash_attn_kernel<<<dim3(SEQ / 64, BATCH * HEADS), 128, 0, stream>>>(
        Qb, Kb, Vtb, A2b);

    // 4) output projection (f32 out)
    wmma_gemm_xwt<2><<<ggrid, 256, 0, stream>>>(A2b, Wob, bo, d_out, 1.0f);
}
